// MPNN_37993280701216
// MI455X (gfx1250) — matrix operand order repaired and checked by reading, not compiled, not run on hardware
//
#include <hip/hip_runtime.h>
#include <hip/hip_bf16.h>

#define N_NODES 50000
#define E_EDGES 250000
#define NG_GRAPHS 2500
#define H_DIM 32

typedef __attribute__((ext_vector_type(16))) __bf16 v16bf;
typedef __attribute__((ext_vector_type(8)))  float  v8f;

static __device__ __forceinline__ v8f wmma_bf16(v16bf a, v16bf b, v8f c) {
  // D = A(16x32 bf16) * B(32x16 bf16) + C(16x16 f32)
  return __builtin_amdgcn_wmma_f32_16x16x32_bf16(false, a, false, b, (short)0, c,
                                                 false, false);
}

// ---- fragment loaders (contiguous k = 16*(lane/16)+i order, same for A and B; wave32) ----
// A 16x32 bf16: lane L holds row m=L&15; lanes<16: K in [0,16), lanes>=16: K in [16,32)
static __device__ __forceinline__ v16bf a_frag_f32(const float* base, int ld, int rows,
                                                   int kmax, int lane) {
  const int m = lane & 15;
  const int k0 = (lane >> 4) << 4;
  v16bf a;
  const float* row = base + m * ld;
  #pragma unroll
  for (int i = 0; i < 8; ++i) {
    int ka = k0 + i, kb = k0 + 8 + i;
    a[i]     = (m < rows && ka < kmax) ? (__bf16)row[ka] : (__bf16)0.0f;
    a[i + 8] = (m < rows && kb < kmax) ? (__bf16)row[kb] : (__bf16)0.0f;
  }
  return a;
}

static __device__ __forceinline__ v16bf a_frag_bf16(const __bf16* base, int ld, int lane) {
  const int m = lane & 15;
  const int k0 = (lane >> 4) << 4;
  const __bf16* row = base + m * ld;
  v16bf a;
  #pragma unroll
  for (int i = 0; i < 8; ++i) { a[i] = row[k0 + i]; a[i + 8] = row[k0 + 8 + i]; }
  return a;
}

// B 32x16 bf16 from f32 row-major [K, ld] (small weights; cached)
static __device__ __forceinline__ v16bf b_frag_f32(const float* base, int ld, int n0,
                                                   int kmax, int lane) {
  const int n = n0 + (lane & 15);
  const int k0 = (lane >> 4) << 4;
  v16bf b;
  #pragma unroll
  for (int i = 0; i < 16; ++i) {
    int k = k0 + i;
    b[i] = (k < kmax) ? (__bf16)base[k * ld + n] : (__bf16)0.0f;
  }
  return b;
}

// B = W.T where W is [outdim, indim] row-major: B[k,n] = W[n, kofs+k]  (per-lane contiguous)
static __device__ __forceinline__ v16bf bT_frag_f32(const float* w, int ld, int n0,
                                                    int kofs, int lane) {
  const int n = n0 + (lane & 15);
  const int k0 = (lane >> 4) << 4;
  v16bf b;
  #pragma unroll
  for (int i = 0; i < 16; ++i) b[i] = (__bf16)w[n * ld + kofs + k0 + i];
  return b;
}

// ---- utility kernels ----
__global__ void k_fillf(float* p, float v, int n) {
  int i = blockIdx.x * blockDim.x + threadIdx.x;
  if (i < n) p[i] = v;
}
__global__ void k_fillu(unsigned* p, unsigned v, int n) {
  int i = blockIdx.x * blockDim.x + threadIdx.x;
  if (i < n) p[i] = v;
}
__global__ void k_deg(const int* ei, float* deg) {
  int e = blockIdx.x * blockDim.x + threadIdx.x;
  if (e < E_EDGES) atomicAdd(&deg[ei[E_EDGES + e]], 1.0f);
}
__global__ void k_denom(float* deg) {
  int i = blockIdx.x * blockDim.x + threadIdx.x;
  if (i < N_NODES) deg[i] = fmaxf(deg[i], 1.0f);
}

// swizzle e_w1 [16,64] f32 -> B-fragment lane-major bf16: [(t*32+l)*16+i], K zero-padded to 32
__global__ void k_swz_ew1(const float* __restrict__ w, __bf16* __restrict__ d) {
  int i0 = blockIdx.x * blockDim.x + threadIdx.x;
  if (i0 >= 4 * 32 * 16) return;
  int i = i0 & 15, l = (i0 >> 4) & 31, t = i0 >> 9;
  int k = ((l >> 4) << 4) + i;
  int n = t * 16 + (l & 15);
  d[i0] = (k < 16) ? (__bf16)w[k * 64 + n] : (__bf16)0.0f;
}
// swizzle e_w2 [64,1024] f32 -> [(t*2+kc)*32+l]*16+i  (t=col-tile, kc=K-chunk)
__global__ void k_swz_ew2(const float* __restrict__ w, __bf16* __restrict__ d) {
  int i0 = blockIdx.x * blockDim.x + threadIdx.x;
  if (i0 >= 64 * 2 * 32 * 16) return;
  int i = i0 & 15, l = (i0 >> 4) & 31, kc = (i0 >> 9) & 1, t = i0 >> 10;
  int k = kc * 32 + ((l >> 4) << 4) + i;
  int n = t * 16 + (l & 15);
  d[i0] = (__bf16)w[k * 1024 + n];
}

// ---- out = relu(x @ lin_w + b); h = out; out_bf = bf16(out) ----
__global__ void k_node_embed(const float* __restrict__ x, const float* __restrict__ lin_w,
                             const float* __restrict__ lin_b, float* __restrict__ out,
                             float* __restrict__ h, __bf16* __restrict__ out_bf) {
  const int lane = threadIdx.x & 31;
  const int tile = (blockIdx.x * blockDim.x + threadIdx.x) >> 5;
  const int row0 = tile * 16;
  if (row0 >= N_NODES) return;
  v16bf a = a_frag_f32(x + (size_t)row0 * 32, 32, 16, 32, lane);
  v8f c0 = {}, c1 = {};
  c0 = wmma_bf16(a, b_frag_f32(lin_w, 32, 0, 32, lane), c0);
  c1 = wmma_bf16(a, b_frag_f32(lin_w, 32, 16, 32, lane), c1);
  const int n = lane & 15, hi = lane >> 4;
  #pragma unroll
  for (int r = 0; r < 8; ++r) {
    int m = row0 + r + 8 * hi;
    float v0 = fmaxf(c0[r] + lin_b[n], 0.0f);
    float v1 = fmaxf(c1[r] + lin_b[16 + n], 0.0f);
    out[(size_t)m * 32 + n] = v0;      out[(size_t)m * 32 + 16 + n] = v1;
    h[(size_t)m * 32 + n] = v0;        h[(size_t)m * 32 + 16 + n] = v1;
    out_bf[(size_t)m * 32 + n] = (__bf16)v0;
    out_bf[(size_t)m * 32 + 16 + n] = (__bf16)v1;
  }
}

// ---- fused edge MLP: W_e = (relu(ea@e_w1+b1) @ e_w2 + b2), stored per-edge in
//      B-fragment lane order: W_e[e*1024 + chunk*512 + lane*16 + i],
//      chunk = o-tile (o>=16), lane = (h>=16)*16 + (o&15), i = h&15 ----
#define EDGEW_WAVES 4
__global__ void k_edgew(const float* __restrict__ edge_attr,
                        const __bf16* __restrict__ ew1_swz, const float* __restrict__ e_b1,
                        const __bf16* __restrict__ ew2_swz, const float* __restrict__ e_b2,
                        __bf16* __restrict__ W_e) {
  __shared__ __bf16 eh_sm[EDGEW_WAVES][16 * 64];   // stage-1 C->A re-layout
  __shared__ __bf16 stage[EDGEW_WAVES][16 * 256];  // stage-2 transpose staging (8KB/wave)
  const int lane = threadIdx.x & 31;
  const int wv = threadIdx.x >> 5;
  const int n = lane & 15, hi = lane >> 4;
  for (int tile = blockIdx.x * EDGEW_WAVES + wv; tile < E_EDGES / 16;
       tile += gridDim.x * EDGEW_WAVES) {
    // stage 1: eh = relu(ea @ e_w1 + b1)  (K=16 zero-padded to 32)
    v16bf a = a_frag_f32(edge_attr + (size_t)tile * 256, 16, 16, 16, lane);
    #pragma unroll
    for (int t = 0; t < 4; ++t) {
      v16bf b = *(const v16bf*)(ew1_swz + ((t * 32 + lane) << 4));
      v8f c = {};
      c = wmma_bf16(a, b, c);
      #pragma unroll
      for (int r = 0; r < 8; ++r) {
        float v = fmaxf(c[r] + e_b1[t * 16 + n], 0.0f);
        eh_sm[wv][(r + 8 * hi) * 64 + t * 16 + n] = (__bf16)v;
      }
    }
    v16bf a0 = a_frag_bf16(&eh_sm[wv][0], 64, lane);
    v16bf a1 = a_frag_bf16(&eh_sm[wv][32], 64, lane);
    // stage 2: 4 groups of 16 col-tiles; each group fills one contiguous 256-elem
    // chunk-slice per edge in consumer layout, then async-copied LDS->global.
    #pragma unroll 1
    for (int g = 0; g < 4; ++g) {
      const int t0 = g & 1, hh = g >> 1;
      #pragma unroll 1
      for (int j = 0; j < 16; ++j) {
        const int t = (hh * 16 + j) * 2 + t0;  // flat output col-tile (h = hh*16+j)
        v8f c = {};
        c = wmma_bf16(a0, *(const v16bf*)(ew2_swz + (((t * 2 + 0) * 32 + lane) << 4)), c);
        c = wmma_bf16(a1, *(const v16bf*)(ew2_swz + (((t * 2 + 1) * 32 + lane) << 4)), c);
        #pragma unroll
        for (int r = 0; r < 8; ++r)
          stage[wv][(r + 8 * hi) * 256 + n * 16 + j] = (__bf16)(c[r] + e_b2[t * 16 + n]);
      }
      // coalesced copy: 16 edges x 512B via async LDS->global (b128 per lane)
      const size_t gbase = (size_t)tile * 16 * 1024 + (size_t)(t0 * 512 + hh * 256);
      unsigned lds0 = (unsigned)(unsigned long long)(size_t)&stage[wv][0];
      asm volatile("s_wait_dscnt 0x0" ::: "memory");
      #pragma unroll
      for (int ii = 0; ii < 8; ++ii) {
        int f = (ii * 32 + lane) * 16;          // flat bf16 index in stage
        int m = f >> 8, wn = f & 255;
        unsigned long long ga =
            (unsigned long long)(size_t)(W_e + gbase + (size_t)m * 1024 + wn);
        unsigned la = lds0 + (unsigned)(f * 2);
        asm volatile("global_store_async_from_lds_b128 %0, %1, off"
                     :: "v"(ga), "v"(la) : "memory");
      }
      asm volatile("s_wait_asynccnt 0x0" ::: "memory");  // before staging reuse
    }
  }
}

// ---- msg_e = out_bf[src] @ W_e[e]; scatter-add to agg[dst] ----
__global__ void k_msg(const int* __restrict__ ei, const __bf16* __restrict__ out_bf,
                      const __bf16* __restrict__ W_e, float* __restrict__ agg) {
  const int lane = threadIdx.x & 31;
  const int stride = (gridDim.x * blockDim.x) >> 5;
  for (int e = (blockIdx.x * blockDim.x + threadIdx.x) >> 5; e < E_EDGES; e += stride) {
    const int s = ei[e];
    const int d = ei[E_EDGES + e];
    const __bf16* wmat = W_e + (size_t)e * 1024;
    if (e + stride < E_EDGES)
      __builtin_prefetch(W_e + (size_t)(e + stride) * 1024, 0, 1);
    v16bf a = {};
    if ((lane & 15) == 0) {  // only row m=0 carries the source feature
      const int k0 = (lane >> 4) << 4;
      #pragma unroll
      for (int i = 0; i < 8; ++i) {
        a[i]     = out_bf[(size_t)s * 32 + k0 + i];
        a[i + 8] = out_bf[(size_t)s * 32 + k0 + 8 + i];
      }
    }
    // W_e is pre-swizzled: one aligned 32B v16bf per lane per fragment
    v16bf b0 = *(const v16bf*)(wmat + (lane << 4));
    v16bf b1 = *(const v16bf*)(wmat + 512 + (lane << 4));
    v8f c0 = {}, c1 = {};
    c0 = wmma_bf16(a, b0, c0);
    c1 = wmma_bf16(a, b1, c1);
    if (lane < 16) {  // row 0 lives in VGPR0 of lanes 0..15
      atomicAdd(&agg[(size_t)d * 32 + lane], c0[0]);
      atomicAdd(&agg[(size_t)d * 32 + 16 + lane], c1[0]);
    }
  }
}

// ---- GRU cell: h = GRU(relu(agg/denom), h); out=h; out_bf=bf16(h) ----
__global__ void k_gru(const float* __restrict__ agg, const float* __restrict__ denom,
                      const float* __restrict__ w_ih, const float* __restrict__ w_hh,
                      const float* __restrict__ b_ih, const float* __restrict__ b_hh,
                      float* __restrict__ h, float* __restrict__ out,
                      __bf16* __restrict__ out_bf) {
  const int lane = threadIdx.x & 31;
  const int tile = (blockIdx.x * blockDim.x + threadIdx.x) >> 5;
  const int row0 = tile * 16;
  if (row0 >= N_NODES) return;
  const int mrow = lane & 15;
  const int ka = (lane >> 4) << 4;
  v16bf am, ah;
  {
    const float* aggr = agg + (size_t)(row0 + mrow) * 32;
    const float dinv = 1.0f / denom[row0 + mrow];
    const float* hr = h + (size_t)(row0 + mrow) * 32;
    #pragma unroll
    for (int i = 0; i < 8; ++i) {
      am[i]     = (__bf16)fmaxf(aggr[ka + i] * dinv, 0.0f);
      am[i + 8] = (__bf16)fmaxf(aggr[ka + 8 + i] * dinv, 0.0f);
      ah[i]     = (__bf16)hr[ka + i];
      ah[i + 8] = (__bf16)hr[ka + 8 + i];
    }
  }
  v8f gi[6], gh[6];
  #pragma unroll
  for (int t = 0; t < 6; ++t) {
    v8f z0 = {}, z1 = {};
    gi[t] = wmma_bf16(am, bT_frag_f32(w_ih, 32, t * 16, 0, lane), z0);
    gh[t] = wmma_bf16(ah, bT_frag_f32(w_hh, 32, t * 16, 0, lane), z1);
  }
  const int n = lane & 15, hi = lane >> 4;
  #pragma unroll
  for (int tp = 0; tp < 2; ++tp) {
    const int col = tp * 16 + n;
    #pragma unroll
    for (int r = 0; r < 8; ++r) {
      const int row = row0 + r + 8 * hi;
      float ir = gi[tp][r] + b_ih[col],          hr_ = gh[tp][r] + b_hh[col];
      float iz = gi[tp + 2][r] + b_ih[32 + col], hz = gh[tp + 2][r] + b_hh[32 + col];
      float in_ = gi[tp + 4][r] + b_ih[64 + col], hn = gh[tp + 4][r] + b_hh[64 + col];
      float rg = 1.0f / (1.0f + __expf(-(ir + hr_)));
      float zg = 1.0f / (1.0f + __expf(-(iz + hz)));
      float ng = tanhf(in_ + rg * hn);
      float hp = h[(size_t)row * 32 + col];
      float hv = (1.0f - zg) * ng + zg * hp;
      h[(size_t)row * 32 + col] = hv;
      out[(size_t)row * 32 + col] = hv;
      out_bf[(size_t)row * 32 + col] = (__bf16)hv;
    }
  }
}

// ---- Set2Set LSTM step (writes hl, cl, and q part of q_star) ----
__global__ void k_lstm(const float* __restrict__ qstar, const float* __restrict__ w_ih,
                       const float* __restrict__ w_hh, const float* __restrict__ b_ih,
                       const float* __restrict__ b_hh, float* __restrict__ hl,
                       float* __restrict__ cl, float* __restrict__ qout) {
  const int lane = threadIdx.x & 31;
  const int tile = (blockIdx.x * blockDim.x + threadIdx.x) >> 5;
  const int row0 = tile * 16;
  if (row0 >= NG_GRAPHS) return;
  const int rows = NG_GRAPHS - row0;
  v16bf aq0 = a_frag_f32(qstar + (size_t)row0 * 64, 64, rows, 32, lane);
  v16bf aq1 = a_frag_f32(qstar + (size_t)row0 * 64 + 32, 64, rows, 32, lane);
  v16bf ah  = a_frag_f32(hl + (size_t)row0 * 32, 32, rows, 32, lane);
  v8f g[8];
  #pragma unroll
  for (int t = 0; t < 8; ++t) {
    v8f z = {};
    z = wmma_bf16(aq0, bT_frag_f32(w_ih, 64, t * 16, 0, lane), z);
    z = wmma_bf16(aq1, bT_frag_f32(w_ih, 64, t * 16, 32, lane), z);
    z = wmma_bf16(ah,  bT_frag_f32(w_hh, 32, t * 16, 0, lane), z);
    g[t] = z;
  }
  const int n = lane & 15, hi = lane >> 4;
  #pragma unroll
  for (int tp = 0; tp < 2; ++tp) {
    const int col = tp * 16 + n;
    #pragma unroll
    for (int r = 0; r < 8; ++r) {
      const int row = row0 + r + 8 * hi;
      if (row >= NG_GRAPHS) continue;
      float gi_ = g[tp][r] + b_ih[col] + b_hh[col];
      float gf_ = g[tp + 2][r] + b_ih[32 + col] + b_hh[32 + col];
      float gg_ = g[tp + 4][r] + b_ih[64 + col] + b_hh[64 + col];
      float go_ = g[tp + 6][r] + b_ih[96 + col] + b_hh[96 + col];
      float ig = 1.0f / (1.0f + __expf(-gi_));
      float fg = 1.0f / (1.0f + __expf(-gf_));
      float gg = tanhf(gg_);
      float og = 1.0f / (1.0f + __expf(-go_));
      float cv = fg * cl[(size_t)row * 32 + col] + ig * gg;
      float hv = og * tanhf(cv);
      cl[(size_t)row * 32 + col] = cv;
      hl[(size_t)row * 32 + col] = hv;
      qout[(size_t)row * 64 + col] = hv;  // q half of q_star
    }
  }
}

// ---- attention scores + segment-max via order-preserving uint ----
__global__ void k_score(const float* __restrict__ out, const float* __restrict__ hl,
                        const int* __restrict__ batch, float* __restrict__ escore,
                        unsigned* __restrict__ emax_ord) {
  int nid = blockIdx.x * blockDim.x + threadIdx.x;
  if (nid >= N_NODES) return;
  int g = batch[nid];
  const float* o = out + (size_t)nid * 32;
  const float* q = hl + (size_t)g * 32;
  float s = 0.0f;
  #pragma unroll
  for (int i = 0; i < 32; ++i) s += o[i] * q[i];
  escore[nid] = s;
  unsigned u = __float_as_uint(s);
  u = (u & 0x80000000u) ? ~u : (u | 0x80000000u);
  atomicMax(&emax_ord[g], u);
}

__global__ void k_expden(const float* __restrict__ escore, const int* __restrict__ batch,
                         const unsigned* __restrict__ emax_ord, float* __restrict__ ex,
                         float* __restrict__ eden) {
  int nid = blockIdx.x * blockDim.x + threadIdx.x;
  if (nid >= N_NODES) return;
  int g = batch[nid];
  unsigned u = emax_ord[g];
  u = (u & 0x80000000u) ? (u & 0x7fffffffu) : ~u;
  float v = __expf(escore[nid] - __uint_as_float(u));
  ex[nid] = v;
  atomicAdd(&eden[g], v);
}

__global__ void k_rpool(const float* __restrict__ out, const int* __restrict__ batch,
                        const float* __restrict__ ex, const float* __restrict__ eden,
                        float* __restrict__ rpool) {
  int lane = threadIdx.x & 31;
  int nid = (blockIdx.x * blockDim.x + threadIdx.x) >> 5;
  if (nid >= N_NODES) return;
  int g = batch[nid];
  float a = ex[nid] / eden[g];
  atomicAdd(&rpool[(size_t)g * 32 + lane], a * out[(size_t)nid * 32 + lane]);
}

__global__ void k_qstar_r(const float* __restrict__ rpool, float* __restrict__ qstar) {
  int i = blockIdx.x * blockDim.x + threadIdx.x;
  if (i >= NG_GRAPHS * 32) return;
  qstar[(size_t)(i >> 5) * 64 + 32 + (i & 31)] = rpool[i];
}

__global__ void k_pred(const float* __restrict__ qstar, const float* __restrict__ pw,
                       const float* __restrict__ pb, float* __restrict__ outp) {
  int g = blockIdx.x * blockDim.x + threadIdx.x;
  if (g >= NG_GRAPHS) return;
  float s = pb[0];
  #pragma unroll
  for (int i = 0; i < 64; ++i) s += qstar[(size_t)g * 64 + i] * pw[i];
  outp[g] = s;
}

extern "C" void kernel_launch(void* const* d_in, const int* in_sizes, int n_in,
                              void* d_out, int out_size, void* d_ws, size_t ws_size,
                              hipStream_t stream) {
  (void)in_sizes; (void)n_in; (void)out_size; (void)ws_size;
  const float* x     = (const float*)d_in[0];
  const int*   ei    = (const int*)d_in[1];
  const float* ea    = (const float*)d_in[2];
  const int*   batch = (const int*)d_in[3];
  const float* lin_w = (const float*)d_in[4];
  const float* lin_b = (const float*)d_in[5];
  const float* e_w1  = (const float*)d_in[6];
  const float* e_b1  = (const float*)d_in[7];
  const float* e_w2  = (const float*)d_in[8];
  const float* e_b2  = (const float*)d_in[9];
  const float* g_wih = (const float*)d_in[10];
  const float* g_whh = (const float*)d_in[11];
  const float* g_bih = (const float*)d_in[12];
  const float* g_bhh = (const float*)d_in[13];
  const float* l_wih = (const float*)d_in[14];
  const float* l_whh = (const float*)d_in[15];
  const float* l_bih = (const float*)d_in[16];
  const float* l_bhh = (const float*)d_in[17];
  const float* p_w   = (const float*)d_in[18];
  const float* p_b   = (const float*)d_in[19];
  float* pred = (float*)d_out;

  char* wsp = (char*)d_ws;
  auto alloc = [&](size_t bytes) -> char* {
    char* p = wsp;
    wsp += (bytes + 255) & ~(size_t)255;
    return p;
  };
  float*  out     = (float*)alloc((size_t)N_NODES * 32 * 4);
  float*  h       = (float*)alloc((size_t)N_NODES * 32 * 4);
  __bf16* out_bf  = (__bf16*)alloc((size_t)N_NODES * 32 * 2);
  float*  agg     = (float*)alloc((size_t)N_NODES * 32 * 4);
  float*  denom   = (float*)alloc((size_t)N_NODES * 4);
  __bf16* ew1_swz = (__bf16*)alloc((size_t)4 * 32 * 16 * 2);
  __bf16* ew2_swz = (__bf16*)alloc((size_t)64 * 2 * 32 * 16 * 2);
  float*  hl      = (float*)alloc((size_t)NG_GRAPHS * 32 * 4);
  float*  cl      = (float*)alloc((size_t)NG_GRAPHS * 32 * 4);
  float*  qstar   = (float*)alloc((size_t)NG_GRAPHS * 64 * 4);
  float*  escore  = (float*)alloc((size_t)N_NODES * 4);
  float*  ex      = (float*)alloc((size_t)N_NODES * 4);
  unsigned* emax  = (unsigned*)alloc((size_t)NG_GRAPHS * 4);
  float*  eden    = (float*)alloc((size_t)NG_GRAPHS * 4);
  float*  rpool   = (float*)alloc((size_t)NG_GRAPHS * 32 * 4);
  __bf16* W_e     = (__bf16*)alloc((size_t)E_EDGES * 1024 * 2);  // 488 MB

  const int BT = 256;
  auto blk = [](long n, int t) { return (int)((n + t - 1) / t); };

  // degree -> denom
  k_fillf<<<blk(N_NODES, BT), BT, 0, stream>>>(denom, 0.0f, N_NODES);
  k_deg<<<blk(E_EDGES, BT), BT, 0, stream>>>(ei, denom);
  k_denom<<<blk(N_NODES, BT), BT, 0, stream>>>(denom);

  // weight swizzles (fragment-lane-major bf16)
  k_swz_ew1<<<blk(4 * 32 * 16, BT), BT, 0, stream>>>(e_w1, ew1_swz);
  k_swz_ew2<<<blk(64 * 2 * 32 * 16, BT), BT, 0, stream>>>(e_w2, ew2_swz);

  // node embedding (one wave per 16-node tile)
  k_node_embed<<<blk((long)(N_NODES / 16) * 32, BT), BT, 0, stream>>>(
      x, lin_w, lin_b, out, h, out_bf);

  // fused edge MLP -> W_e (consumer layout), 4 waves/block, 40KB LDS
  k_edgew<<<blk(E_EDGES / 16, EDGEW_WAVES), EDGEW_WAVES * 32, 0, stream>>>(
      ea, ew1_swz, e_b1, ew2_swz, e_b2, W_e);

  // 3 message-passing steps
  for (int step = 0; step < 3; ++step) {
    k_fillf<<<blk((long)N_NODES * 32, BT), BT, 0, stream>>>(agg, 0.0f, N_NODES * 32);
    k_msg<<<4096, BT, 0, stream>>>(ei, out_bf, W_e, agg);
    k_gru<<<blk((long)(N_NODES / 16) * 32, BT), BT, 0, stream>>>(
        agg, denom, g_wih, g_whh, g_bih, g_bhh, h, out, out_bf);
  }

  // Set2Set init
  k_fillf<<<blk((long)NG_GRAPHS * 64, BT), BT, 0, stream>>>(qstar, 0.0f, NG_GRAPHS * 64);
  k_fillf<<<blk((long)NG_GRAPHS * 32, BT), BT, 0, stream>>>(hl, 0.0f, NG_GRAPHS * 32);
  k_fillf<<<blk((long)NG_GRAPHS * 32, BT), BT, 0, stream>>>(cl, 0.0f, NG_GRAPHS * 32);

  const int ltiles = (NG_GRAPHS + 15) / 16;
  for (int step = 0; step < 3; ++step) {
    k_lstm<<<blk((long)ltiles * 32, BT), BT, 0, stream>>>(
        qstar, l_wih, l_whh, l_bih, l_bhh, hl, cl, qstar);
    k_fillu<<<blk(NG_GRAPHS, BT), BT, 0, stream>>>(emax, 0u, NG_GRAPHS);
    k_score<<<blk(N_NODES, BT), BT, 0, stream>>>(out, hl, batch, escore, emax);
    k_fillf<<<blk(NG_GRAPHS, BT), BT, 0, stream>>>(eden, 0.0f, NG_GRAPHS);
    k_expden<<<blk(N_NODES, BT), BT, 0, stream>>>(escore, batch, emax, ex, eden);
    k_fillf<<<blk((long)NG_GRAPHS * 32, BT), BT, 0, stream>>>(rpool, 0.0f, NG_GRAPHS * 32);
    k_rpool<<<blk((long)N_NODES * 32, BT), BT, 0, stream>>>(out, batch, ex, eden, rpool);
    k_qstar_r<<<blk((long)NG_GRAPHS * 32, BT), BT, 0, stream>>>(rpool, qstar);
  }

  k_pred<<<blk(NG_GRAPHS, BT), BT, 0, stream>>>(qstar, p_w, p_b, pred);
}
